// Trittention_17875653886376
// MI455X (gfx1250) — matrix operand order repaired and checked by reading, not compiled, not run on hardware
//
#include <hip/hip_runtime.h>
#include <hip/hip_bf16.h>
#include <math.h>

// ---------------------------------------------------------------------------
// Trittention for MI455X (gfx1250, wave32, WMMA + TDM).
// Factorized softmax:  attn[s,t,q] = Eqk[t,q]*Ekk[s,t]/Z(q)  on s<=q, t<=q.
// Chunked-prefix algorithm, one workgroup per (b, head, 16-wide q-chunk).
// GEMMs via v_wmma_f32_16x16x32_f16 (f16 in, f32 accumulate); all WMMA
// fragments are contiguous LDS reads (2x ds_load_b128), enabled by
// pre-transposed layouts (WT[c][k] weights, SV1/V2 as [h][t], Ekk as [t][s]).
// Bulk global->LDS tile staging goes through the Tensor Data Mover
// (tensor_load_to_lds + s_wait_tensorcnt) when the builtin is available.
// Workspace: 6*196608 f16 (mats+z) + 3072*512 f16 (WT) = 5,505,024 bytes.
// ---------------------------------------------------------------------------

typedef __attribute__((ext_vector_type(16))) _Float16 v16h;
typedef __attribute__((ext_vector_type(8)))  _Float16 h8;
typedef __attribute__((ext_vector_type(4)))  _Float16 h4;
typedef __attribute__((ext_vector_type(8)))  float    v8f;
typedef __attribute__((ext_vector_type(4)))  float    f4;
typedef __attribute__((ext_vector_type(4)))  unsigned int u32x4;
typedef __attribute__((ext_vector_type(8)))  int      i32x8;
typedef __attribute__((ext_vector_type(4)))  int      i32x4;

#define T_SEQ   192
#define B_SZ    2
#define NH      8
#define DH      64
#define DMODEL  512
#define BT      (B_SZ * T_SEQ)             /* 384 */
#define MAT_ELEMS (B_SZ * NH * T_SEQ * DH) /* 196608 f16 per matrix */
#define NCHUNK  (T_SEQ / 16)               /* 12 */
#define WT_OFF  (6 * (size_t)MAT_ELEMS)    /* WT location in ws (halfs) */

#if __has_builtin(__builtin_amdgcn_tensor_load_to_lds) && \
    __has_builtin(__builtin_amdgcn_s_wait_tensorcnt)
#define USE_TDM 1
#if __has_include(<hip/amd_detail/amd_gfx1250_TDM.h>)
#define TDM_SIX_ARGS 1
#endif
#endif

__device__ __forceinline__ v8f wmma_f16(v16h a, v16h b, v8f c) {
  return __builtin_amdgcn_wmma_f32_16x16x32_f16(false, a, false, b, (short)0, c,
                                                false, false);
}

// A-fragment (16x32 f16): lane holds K = kb+8*lhi+[0,8) and +16. 2x b128.
__device__ __forceinline__ v16h ld_frag_a(const _Float16* row, int kb, int lhi) {
  const h8* p = (const h8*)(row + kb + 8 * lhi);
  h8 lo = p[0], hi = p[2];
  v16h a;
#pragma unroll
  for (int e = 0; e < 8; ++e) { a[e] = lo[e]; a[e + 8] = hi[e]; }
  return a;
}

// B-fragment where B[k][n] = row_n[k]: K = kb+8*lhi+[0,8) and +16. 2x b128.
__device__ __forceinline__ v16h ld_frag_bt(const _Float16* row, int kb, int lhi) {
  const h8* p = (const h8*)(row + kb + 8 * lhi);
  h8 lo = p[0], hi = p[2];
  v16h b;
#pragma unroll
  for (int e = 0; e < 8; ++e) { b[e] = lo[e]; b[e + 8] = hi[e]; }
  return b;
}

#ifdef USE_TDM
// TDM 2D tile load (f16 elements): rows x tile_w tile from a row-major global
// tensor with row stride `row_stride`, packed contiguously into LDS.
// D# built per CDNA5 ISA 8.3/8.4 (group0: count/lds/global/type=2;
// group1: data_size=2B, dims, tile dims, dim0 stride; groups 2/3 zero).
__device__ __forceinline__ void tdm_load_2d(const void* g, void* l,
                                            unsigned tile_w, unsigned rows,
                                            unsigned row_stride) {
  const unsigned long long ga = (unsigned long long)g;
  u32x4 g0;
  g0[0] = 1u;                                          // count=1 (valid, user)
  g0[1] = (unsigned)(unsigned long long)l;             // lds_addr (low 32 bits)
  g0[2] = (unsigned)ga;                                // global_addr[31:0]
  g0[3] = (unsigned)((ga >> 32) & 0x01FFFFFFu) | (2u << 30); // addr[56:32]|type=2
  i32x8 g1;
  g1[0] = (int)(1u << 16);                             // mask=0, data_size=2B
  g1[1] = (int)((row_stride & 0xFFFFu) << 16);         // tensor_dim0[15:0]
  g1[2] = (int)((row_stride >> 16) | ((rows & 0xFFFFu) << 16)); // tdim0 hi|tdim1 lo
  g1[3] = (int)((rows >> 16) | (tile_w << 16));        // tdim1 hi | tile_dim0
  g1[4] = (int)rows;                                   // tile_dim1, tile_dim2=0
  g1[5] = (int)row_stride;                             // tensor_dim0_stride lo
  g1[6] = 0; g1[7] = 0;
  const i32x4 zz = {0, 0, 0, 0};
#ifdef TDM_SIX_ARGS
  const i32x8 z8 = {0, 0, 0, 0, 0, 0, 0, 0};
  __builtin_amdgcn_tensor_load_to_lds(g0, g1, zz, zz, z8, 0);
#else
  __builtin_amdgcn_tensor_load_to_lds(g0, g1, zz, zz, 0);
#endif
}
#endif

// ---------------------------------------------------------------------------
// Kernel 0: transpose+convert all weights to f16  WT[c][k], c in [0,3072):
//   c<1536: Wkkq | c<2048: Wv1 | c<2560: Wv2 | else: Wout.
// ---------------------------------------------------------------------------
__global__ __launch_bounds__(256)
void wt_kernel(const float* __restrict__ Wkkq, const float* __restrict__ Wv1,
               const float* __restrict__ Wv2,  const float* __restrict__ Wout,
               _Float16* __restrict__ WT) {
  const int idx = blockIdx.x * 256 + threadIdx.x;   // 196608 threads
  const int c   = idx >> 6;
  const int k8  = (idx & 63) << 3;
  const float* W; int wcols, cc;
  if (c < 1536)      { W = Wkkq; wcols = 1536; cc = c; }
  else if (c < 2048) { W = Wv1;  wcols = 512;  cc = c - 1536; }
  else if (c < 2560) { W = Wv2;  wcols = 512;  cc = c - 2048; }
  else               { W = Wout; wcols = 512;  cc = c - 2560; }
  h8 v;
#pragma unroll
  for (int j = 0; j < 8; ++j)
    v[j] = (_Float16)W[(size_t)(k8 + j) * wcols + cc];
  *(h8*)&WT[(size_t)c * DMODEL + k8] = v;
}

// ---------------------------------------------------------------------------
// Kernel 1: fused projections. O[384 x 2560] = x @ WT^T + bias, silu on v1.
// Block = 16 rows x 512 cols (8 waves x 64 cols); grid 24 x 5.
// K1/K2/Q written [t][h]; SV1/V2 written TRANSPOSED [h][t] per head.
// ---------------------------------------------------------------------------
__global__ __launch_bounds__(256)
void proj_kernel(const float* __restrict__ x, const _Float16* __restrict__ WT,
                 const float* __restrict__ bkkq, const float* __restrict__ bv1,
                 const float* __restrict__ bv2, _Float16* __restrict__ ws) {
  const int tid  = threadIdx.x;
  const int lane = tid & 31, lm = lane & 15, lhi = lane >> 4;
  const int wv   = tid >> 5;
  const int mt   = blockIdx.x / 5;
  const int g    = blockIdx.x % 5;
  const int r0   = mt * 16;
  const int c0   = g * 512;

  __shared__ _Float16 sX [16 * DMODEL];      // 16 KB activation slab (f16)
  __shared__ _Float16 sWs[512 * 32];         // 32 KB weight slab WT[c][32k]

  for (int i = tid; i < 16 * DMODEL / 4; i += 256) {   // stage x once (f32->f16)
    const int row = i >> 7, c4 = (i & 127) << 2;
    f4 v = *(const f4*)&x[(size_t)(r0 + row) * DMODEL + c4];
    h4 hv; hv[0] = (_Float16)v[0]; hv[1] = (_Float16)v[1];
    hv[2] = (_Float16)v[2]; hv[3] = (_Float16)v[3];
    *(h4*)&sX[row * DMODEL + c4] = hv;
  }

  v8f acc[4] = { {}, {}, {}, {} };
  for (int kb = 0; kb < DMODEL; kb += 32) {
    __syncthreads();
#ifdef USE_TDM
    if (tid < 32) {                          // wave 0: TDM 512x32 f16 slab
      tdm_load_2d(&WT[(size_t)c0 * DMODEL + kb], sWs, 32, 512, DMODEL);
      __builtin_amdgcn_s_wait_tensorcnt(0);
    }
#else
    for (int i = tid; i < 512 * 4; i += 256) {
      const int c = i >> 2, jj = i & 3;
      const h8* src = (const h8*)&WT[(size_t)(c0 + c) * DMODEL + kb];
      *(h8*)&sWs[c * 32 + jj * 8] = src[jj];
    }
#endif
    __syncthreads();

    v16h a = ld_frag_a(&sX[lm * DMODEL], kb, lhi);
#pragma unroll
    for (int nt = 0; nt < 4; ++nt) {
      const int clocal = wv * 64 + nt * 16 + lm;
      acc[nt] = wmma_f16(a, ld_frag_bt(&sWs[clocal * 32], 0, lhi), acc[nt]);
    }
  }

#pragma unroll
  for (int nt = 0; nt < 4; ++nt) {
    const int c = c0 + wv * 64 + nt * 16 + lm;
    int mat, cc; float bias;
    if (c < 512)       { mat = 0; cc = c;        bias = bkkq[c]; }
    else if (c < 1024) { mat = 1; cc = c - 512;  bias = bkkq[c]; }
    else if (c < 1536) { mat = 2; cc = c - 1024; bias = bkkq[c]; }
    else if (c < 2048) { mat = 3; cc = c - 1536; bias = bv1[cc]; }
    else               { mat = 4; cc = c - 2048; bias = bv2[cc]; }
    const int n = cc >> 6, h = cc & 63;
#pragma unroll
    for (int r = 0; r < 8; ++r) {
      const int row = r0 + r + 8 * lhi;
      const int b = row / T_SEQ, t = row % T_SEQ;
      float v = acc[nt][r] + bias;
      if (mat == 3) v = v / (1.0f + expf(-v));          // silu(v1)
      const size_t hb = (size_t)(b * NH + n) * T_SEQ * DH;
      const size_t off = (mat >= 3) ? (hb + (size_t)h * T_SEQ + t)   // [h][t]
                                    : (hb + (size_t)t * DH + h);     // [t][h]
      ws[(size_t)mat * MAT_ELEMS + off] = (_Float16)v;
    }
  }
}

// ---------------------------------------------------------------------------
// Kernel 2: chunked trittention core. 192 blocks = (b,n) x 12 q-chunks.
// 384 threads = 12 waves; wave w owns t-tile w. Prefix A[t,h] in registers.
// Dynamic LDS ~131 KB (within 320 KB WGP LDS).
// ---------------------------------------------------------------------------
#define ATTN_SMEM_BYTES \
  ((16 * T_SEQ + T_SEQ + T_SEQ + 16 * DH + 16) * 4 + \
   (4 * T_SEQ * DH + 16 * DH + 32 * T_SEQ) * 2)

__global__ __launch_bounds__(384)
void attn_kernel(const _Float16* __restrict__ ws_in, _Float16* __restrict__ zbuf) {
  const int tid  = threadIdx.x;
  const int lane = tid & 31, lm = lane & 15, lhi = lane >> 4;
  const int w    = tid >> 5;                  // wave id 0..11 == t-tile
  const int ci   = blockIdx.x % NCHUNK;
  const int bn   = blockIdx.x / NCHUNK;
  const int b    = bn >> 3, n = bn & 7;
  const int q0   = ci * 16;
  const int tmax = q0 + 16;

  const size_t base = (size_t)bn * T_SEQ * DH;
  const _Float16* K1g   = ws_in + 0 * (size_t)MAT_ELEMS + base;  // [t][h]
  const _Float16* K2g   = ws_in + 1 * (size_t)MAT_ELEMS + base;  // [t][h]
  const _Float16* Qg    = ws_in + 2 * (size_t)MAT_ELEMS + base;  // [t][h]
  const _Float16* SV1gT = ws_in + 3 * (size_t)MAT_ELEMS + base;  // [h][t]
  const _Float16* V2gT  = ws_in + 4 * (size_t)MAT_ELEMS + base;  // [h][t]

  extern __shared__ __align__(16) char smem_raw[];
  float* sEqk = (float*)smem_raw;             // [16][192] Eqk^T (q-major)
  float* sW   = sEqk + 16 * T_SEQ;            // [192]
  float* sRow = sW + T_SEQ;                   // [192]
  float* sN   = sRow + T_SEQ;                 // [16][64]
  float* sZ   = sN + 16 * DH;                 // [16]
  _Float16* sK1   = (_Float16*)(sZ + 16);     // [192][64]
  _Float16* sK2   = sK1 + T_SEQ * DH;         // [192][64]
  _Float16* sSV1T = sK2 + T_SEQ * DH;         // [64][192]
  _Float16* sV2T  = sSV1T + T_SEQ * DH;       // [64][192]
  _Float16* sQ    = sV2T + T_SEQ * DH;        // [16][64]
  _Float16* sEkkT = sQ + 16 * DH;             // [192][32]  Ekk^T (t-major)

#ifdef USE_TDM
  if (tid < 32) {                             // wave 0 issues 5 TDM tile loads
    tdm_load_2d(K1g,  sK1,  DH, T_SEQ, DH);
    tdm_load_2d(K2g,  sK2,  DH, T_SEQ, DH);
    tdm_load_2d(SV1gT, sSV1T, T_SEQ, DH, T_SEQ);
    tdm_load_2d(V2gT,  sV2T,  T_SEQ, DH, T_SEQ);
    tdm_load_2d(Qg + (size_t)q0 * DH, sQ, DH, 16, DH);
    __builtin_amdgcn_s_wait_tensorcnt(0);
  }
#else
  {                                           // b128 block copies into LDS
    const h8* g0 = (const h8*)K1g;   h8* d0 = (h8*)sK1;
    const h8* g1 = (const h8*)K2g;   h8* d1 = (h8*)sK2;
    const h8* g2 = (const h8*)SV1gT; h8* d2 = (h8*)sSV1T;
    const h8* g3 = (const h8*)V2gT;  h8* d3 = (h8*)sV2T;
    for (int i = tid; i < T_SEQ * DH / 8; i += 384) {
      d0[i] = g0[i]; d1[i] = g1[i]; d2[i] = g2[i]; d3[i] = g3[i];
    }
    const h8* gq = (const h8*)(Qg + (size_t)q0 * DH); h8* dq = (h8*)sQ;
    for (int i = tid; i < 16 * DH / 8; i += 384) dq[i] = gq[i];
  }
#endif
  for (int i = tid; i < 16 * DH; i += 384) sN[i] = 0.f;
  if (tid < T_SEQ) sW[tid] = 0.f;
  if (tid < 16)    sZ[tid] = 0.f;
  __syncthreads();

  // ---- Phase 1: Eqk[t,q] = exp(qk/64) masked (t<=q) ------------------------
  if (w <= ci) {
    v8f c = {};
#pragma unroll
    for (int kb = 0; kb < DH; kb += 32)
      c = wmma_f16(ld_frag_a(&sQ[lm * DH], kb, lhi),
                   ld_frag_bt(&sK2[(16 * w + lm) * DH], kb, lhi), c);
#pragma unroll
    for (int r = 0; r < 8; ++r) {
      const int ql = r + 8 * lhi;
      const int t  = 16 * w + lm;
      sEqk[ql * T_SEQ + t] = (t <= q0 + ql) ? expf(c[r] * 0.015625f) : 0.f;
    }
  }
  __syncthreads();

  // ---- Phase 2: prefix A[t,h] = sum_{s<q0} Ekk[s,t]*sv1[s,h],  W[t] --------
  v8f Aacc[4] = { {}, {}, {}, {} };
  for (int ss = 0; ss < q0; ss += 32) {
    if (w <= ci) {
#pragma unroll
      for (int sub = 0; sub < 2; ++sub) {
        const int sbase = ss + 16 * sub;
        h8 ev = {};
        if (sbase < q0) {
          v8f c = {};
#pragma unroll
          for (int kb = 0; kb < DH; kb += 32)
            c = wmma_f16(ld_frag_a(&sK1[(sbase + lm) * DH], kb, lhi),
                         ld_frag_bt(&sK2[(16 * w + lm) * DH], kb, lhi), c);
#pragma unroll
          for (int r = 0; r < 8; ++r) ev[r] = (_Float16)expf(c[r] * 0.015625f);
        }
        // lane's 8 s-values are contiguous in the [t][s] layout: one b128
        *(h8*)&sEkkT[(16 * w + lm) * 32 + 16 * sub + 8 * lhi] = ev;
      }
    }
    __syncthreads();
    if (tid < tmax) {                          // W[t] += row sums (vector)
      const h8* p = (const h8*)&sEkkT[tid * 32];
      float acc = sW[tid];
#pragma unroll
      for (int kk = 0; kk < 4; ++kk) {
        h8 v = p[kk];
#pragma unroll
        for (int e = 0; e < 8; ++e) acc += (float)v[e];
      }
      sW[tid] = acc;
    }
    if (w <= ci) {                             // A += Ekk^T @ SV1  (K = 32 s)
      v16h a = ld_frag_a(&sEkkT[(16 * w + lm) * 32], 0, lhi);
#pragma unroll
      for (int ht = 0; ht < 4; ++ht) {
        v16h bf = ld_frag_bt(&sSV1T[(16 * ht + lm) * T_SEQ], ss, lhi);
        Aacc[ht] = wmma_f16(a, bf, Aacc[ht]);
      }
    }
    __syncthreads();                           // before sEkkT is overwritten
  }

  // ---- Phase 3: intra-chunk sequential over q = q0+j (s = q row added) -----
  for (int j = 0; j < 16; ++j) {
    const int q = q0 + j;
    if (tid < tmax) {                          // ekk row for s = q (vector dot)
      float dot = 0.f;
      const h8* k1p = (const h8*)&sK1[q * DH];
      const h8* k2p = (const h8*)&sK2[tid * DH];
#pragma unroll
      for (int kk = 0; kk < 8; ++kk) {
        h8 av = k1p[kk], bv = k2p[kk];
#pragma unroll
        for (int e = 0; e < 8; ++e) dot += (float)av[e] * (float)bv[e];
      }
      const float v = expf(dot * 0.015625f);
      sRow[tid] = v;
      sW[tid]  += v;                           // W now covers s <= q
    }
    __syncthreads();
    if (w <= ci) {
      const int tb = 16 * w + 8 * lhi;         // lane's 8 consecutive t
      const f4* rp = (const f4*)&sRow[tb];
      const f4* ep = (const f4*)&sEqk[j * T_SEQ + tb];
      f4 rv0 = rp[0], rv1 = rp[1];
      f4 ev0 = ep[0], ev1 = ep[1];
#pragma unroll
      for (int ht = 0; ht < 4; ++ht) {
        const int h  = 16 * ht + lm;
        const float sv = (float)sSV1T[h * T_SEQ + q];
        h8 v2v = *(const h8*)&sV2T[h * T_SEQ + tb];
        float partial = 0.f;
#pragma unroll
        for (int r = 0; r < 8; ++r) {
          const float rv = (r < 4) ? rv0[r] : rv1[r - 4];
          const float evv = (r < 4) ? ev0[r] : ev1[r - 4];
          Aacc[ht][r] += rv * sv;              // rank-1 update (s = q)
          partial += evv * (float)v2v[r] * Aacc[ht][r];
        }
        atomicAdd(&sN[j * DH + h], partial);   // ds_add_f32
      }
    }
    if (tid < tmax)                            // Z(q) = sum_t Eqk * W
      atomicAdd(&sZ[j], sEqk[j * T_SEQ + tid] * sW[tid]);
    __syncthreads();
  }

  // ---- write z[q, n, h] = N / Z as f16 rows of [BT x 512] ------------------
  for (int i = tid; i < 16 * DH; i += 384) {
    const int j = i >> 6, h = i & 63;
    zbuf[((size_t)b * T_SEQ + q0 + j) * DMODEL + n * DH + h] =
        (_Float16)(sN[i] / sZ[j]);
  }
}

// ---------------------------------------------------------------------------
// Kernel 3: out = Z[384x512] @ Wout + bout (f32 out), Wout via WT cols 2560+.
// ---------------------------------------------------------------------------
__global__ __launch_bounds__(256)
void outproj_kernel(const _Float16* __restrict__ zbuf,
                    const _Float16* __restrict__ WT,
                    const float* __restrict__ bout,
                    float* __restrict__ out) {
  const int tid  = threadIdx.x;
  const int lane = tid & 31, lm = lane & 15, lhi = lane >> 4;
  const int wv   = tid >> 5;
  const int r0   = blockIdx.x * 16;

  __shared__ _Float16 sZt[16 * DMODEL];       // 16 KB
  __shared__ _Float16 sWs[512 * 32];          // 32 KB

#ifdef USE_TDM
  if (tid < 32) {
    tdm_load_2d(zbuf + (size_t)r0 * DMODEL, sZt, DMODEL, 16, DMODEL);
    __builtin_amdgcn_s_wait_tensorcnt(0);
  }
#else
  {
    const h8* gz = (const h8*)(zbuf + (size_t)r0 * DMODEL);
    h8* dz = (h8*)sZt;
    for (int i = tid; i < 16 * DMODEL / 8; i += 256) dz[i] = gz[i];
  }
#endif

  v8f acc[4] = { {}, {}, {}, {} };
  for (int kb = 0; kb < DMODEL; kb += 32) {
    __syncthreads();
#ifdef USE_TDM
    if (tid < 32) {
      tdm_load_2d(&WT[(size_t)2560 * DMODEL + kb], sWs, 32, 512, DMODEL);
      __builtin_amdgcn_s_wait_tensorcnt(0);
    }
#else
    for (int i = tid; i < 512 * 4; i += 256) {
      const int c = i >> 2, jj = i & 3;
      const h8* src = (const h8*)&WT[(size_t)(2560 + c) * DMODEL + kb];
      *(h8*)&sWs[c * 32 + jj * 8] = src[jj];
    }
#endif
    __syncthreads();

    v16h a = ld_frag_a(&sZt[lm * DMODEL], kb, lhi);
#pragma unroll
    for (int nt = 0; nt < 4; ++nt) {
      const int c = wv * 64 + nt * 16 + lm;
      acc[nt] = wmma_f16(a, ld_frag_bt(&sWs[c * 32], 0, lhi), acc[nt]);
    }
  }
#pragma unroll
  for (int nt = 0; nt < 4; ++nt) {
    const int c = wv * 64 + nt * 16 + lm;
    const float bias = bout[c];
#pragma unroll
    for (int r = 0; r < 8; ++r)
      out[(size_t)(r0 + r + 8 * lhi) * DMODEL + c] = acc[nt][r] + bias;
  }
}

// ---------------------------------------------------------------------------
extern "C" void kernel_launch(void* const* d_in, const int* in_sizes, int n_in,
                              void* d_out, int out_size, void* d_ws, size_t ws_size,
                              hipStream_t stream) {
  (void)in_sizes; (void)n_in; (void)out_size; (void)ws_size;
  const float* x    = (const float*)d_in[0];
  const float* Wkkq = (const float*)d_in[1];
  const float* bkkq = (const float*)d_in[2];
  const float* Wv1  = (const float*)d_in[3];
  const float* bv1  = (const float*)d_in[4];
  const float* Wv2  = (const float*)d_in[5];
  const float* bv2  = (const float*)d_in[6];
  const float* Wout = (const float*)d_in[7];
  const float* bout = (const float*)d_in[8];
  _Float16* ws  = (_Float16*)d_ws;            // needs ~5.3 MB
  float*    out = (float*)d_out;
  _Float16* WT  = ws + WT_OFF;
  _Float16* zb  = ws + 5 * (size_t)MAT_ELEMS;

  wt_kernel<<<768, 256, 0, stream>>>(Wkkq, Wv1, Wv2, Wout, WT);
  proj_kernel<<<120, 256, 0, stream>>>(x, WT, bkkq, bv1, bv2, ws);
  attn_kernel<<<B_SZ * NH * NCHUNK, 384, ATTN_SMEM_BYTES, stream>>>(ws, zb);
  outproj_kernel<<<24, 256, 0, stream>>>(zb, WT, bout, out);
}
